// SingularAttention_86844238725657
// MI455X (gfx1250) — matrix operand order repaired and checked by reading, not compiled, not run on hardware
//
#include <hip/hip_runtime.h>
#include <hip/hip_bf16.h>

// ---------------------------------------------------------------------------
// Problem constants (from reference): B=8, L=2048, C=512, R=256, H=8, DK=64
// ---------------------------------------------------------------------------
static constexpr int cB  = 8;
static constexpr int cL  = 2048;
static constexpr int cC  = 512;
static constexpr int cR  = 256;
static constexpr int cH  = 8;
static constexpr int cDK = 64;
static constexpr int cHD = cH * cDK;           // 512
static constexpr float MASK_NEG_F = -32767.0f; // -2^15 + 1

typedef __bf16 bf16_t;
typedef __attribute__((ext_vector_type(16))) __bf16 v16bf;
typedef __attribute__((ext_vector_type(8)))  __bf16 v8bf;
typedef __attribute__((ext_vector_type(8)))  float  v8f;

// ---------------------------------------------------------------------------
// CDNA5 async Global->LDS copy (ASYNCcnt path), with sync fallback.
// Probe round 2: builtin exists; arg0 must be non-const
// 'int __attribute__((vector_size(16)))' in AS1 (printed as __device__).
// ---------------------------------------------------------------------------
#if defined(__AMDGCN__) && __has_builtin(__builtin_amdgcn_global_load_async_to_lds_b128) && __has_builtin(__builtin_amdgcn_s_wait_asynccnt)
#define ATHENA_ASYNC_LDS 1
#else
#define ATHENA_ASYNC_LDS 0
#endif

__device__ __forceinline__ void copy16_g2l(const bf16_t* __restrict__ g, bf16_t* l)
{
#if ATHENA_ASYNC_LDS
    typedef int v4i_b128 __attribute__((vector_size(16)));
    __builtin_amdgcn_global_load_async_to_lds_b128(
        (__attribute__((address_space(1))) v4i_b128*)g,
        (__attribute__((address_space(3))) v4i_b128*)l, 0, 0);
#else
    *(v8bf*)l = *(const v8bf*)g;
#endif
}

__device__ __forceinline__ void async_join()
{
#if ATHENA_ASYNC_LDS
    __builtin_amdgcn_s_wait_asynccnt(0);
#endif
}

// ---------------------------------------------------------------------------
// Block reductions (blockDim.x == 256)
// ---------------------------------------------------------------------------
__device__ __forceinline__ float block_max256(float v, float* red) {
    const int t = threadIdx.x;
    red[t] = v; __syncthreads();
    #pragma unroll
    for (int off = 128; off >= 1; off >>= 1) {
        if (t < off) red[t] = fmaxf(red[t], red[t + off]);
        __syncthreads();
    }
    float r = red[0]; __syncthreads();
    return r;
}

__device__ __forceinline__ float block_sum256(float v, float* red) {
    const int t = threadIdx.x;
    red[t] = v; __syncthreads();
    #pragma unroll
    for (int off = 128; off >= 1; off >>= 1) {
        if (t < off) red[t] = red[t] + red[t + off];
        __syncthreads();
    }
    float r = red[0]; __syncthreads();
    return r;
}

// ---------------------------------------------------------------------------
// f32 -> bf16 conversion (vectorized; n is always a multiple of 4)
// ---------------------------------------------------------------------------
__global__ __launch_bounds__(256) void f32_to_bf16_kernel(
    const float* __restrict__ in, bf16_t* __restrict__ out, long long n)
{
    long long i = ((long long)blockIdx.x * blockDim.x + threadIdx.x) * 4;
    if (i + 3 < n) {
        float4 f = *(const float4*)(in + i);
        out[i + 0] = (bf16_t)f.x;
        out[i + 1] = (bf16_t)f.y;
        out[i + 2] = (bf16_t)f.z;
        out[i + 3] = (bf16_t)f.w;
    }
}

// ---------------------------------------------------------------------------
// Generic batched bf16 GEMM, f32 WMMA accumulation, double-buffered LDS.
//   C[M,N] (f32 or bf16, *scale) = A[M,K] x B[K,N]
//   B_NT=true : B supplied as [N,K] row-major (weights / gram / Q.K^T)
//   B_NT=false: B supplied as [K,N] row-major (activations)
// Block: 256 threads = 8 waves; tile 128x64x32.  Waves in a 4x2 grid; each
// wave owns a 32x32 sub-tile = 2x2 accumulators of v_wmma_f32_16x16x32_bf16
// (2 A-frags + 2 B-frags per K-step -> 2 ds_load_b128 per WMMA).
// Next K-tile is staged (async Global->LDS where available) into the ping-
// pong buffer while the current tile's WMMAs run; one barrier per K-step.
// ---------------------------------------------------------------------------
static constexpr int TLM = 128, TLN = 64, TLK = 32, LDSS = 40; // bf16 elems

__device__ __forceinline__ v16bf frag_a(const bf16_t* Ls, int row0, int lane)
{
    // 16-bit A 16x32 layout: lanes 0-15 hold K {0..7,16..23}, 16-31 {8..15,24..31}
    const int row = row0 + (lane & 15);
    const int g   = (lane >> 4) << 3;
    v8bf lo = *(const v8bf*)(Ls + row * LDSS + g);
    v8bf hi = *(const v8bf*)(Ls + row * LDSS + 16 + g);
    return __builtin_shufflevector(lo, hi,
        0, 1, 2, 3, 4, 5, 6, 7, 8, 9, 10, 11, 12, 13, 14, 15);
}

__device__ __forceinline__ v16bf frag_b(const bf16_t* Ls, int col0, int lane)
{
    // 16-bit B 32x16 layout: lane = column N; lanes 0-15 K {0..7,16..23}, 16-31 {8..15,24..31}
    const int col = col0 + (lane & 15);
    const int g   = (lane >> 4) << 3;
    v8bf lo = *(const v8bf*)(Ls + col * LDSS + g);
    v8bf hi = *(const v8bf*)(Ls + col * LDSS + 16 + g);
    return __builtin_shufflevector(lo, hi,
        0, 1, 2, 3, 4, 5, 6, 7, 8, 9, 10, 11, 12, 13, 14, 15);
}

template <bool B_NT, bool OUT_BF16>
__global__ __launch_bounds__(256) void gemm_bf16_kernel(
    const bf16_t* __restrict__ A, const bf16_t* __restrict__ Bm,
    void* __restrict__ Cout,
    int K, int lda, int ldb, int ldc,
    long long sAo, long long sAi, long long sBo, long long sBi,
    long long sCo, long long sCi, int zdiv, float scale)
{
    __shared__ __align__(16) bf16_t As[2][TLM * LDSS];
    __shared__ __align__(16) bf16_t Bs[2][TLN * LDSS];

    const int zo = blockIdx.z / zdiv;
    const int zi = blockIdx.z - zo * zdiv;
    A  += (long long)zo * sAo + (long long)zi * sAi;
    Bm += (long long)zo * sBo + (long long)zi * sBi;

    const int tid  = threadIdx.x;
    const int wave = tid >> 5;
    const int lane = tid & 31;
    const int m0 = blockIdx.x * TLM;
    const int n0 = blockIdx.y * TLN;
    const int wm = (wave >> 1) * 32;   // wave row:  0,32,64,96
    const int wn = (wave & 1) * 32;    // wave col:  0,32

    v8f acc[2][2];
    #pragma unroll
    for (int i = 0; i < 2; ++i)
        #pragma unroll
        for (int j = 0; j < 2; ++j)
            #pragma unroll
            for (int e = 0; e < 8; ++e) acc[i][j][e] = 0.0f;

    auto stage = [&](int buf, int k0) {
        // A tile 128x32: 512 chunks of 8 bf16, 2 per thread
        #pragma unroll
        for (int c = 0; c < 2; ++c) {
            const int chunk = tid + c * 256;
            const int row = chunk >> 2;
            const int col = (chunk & 3) << 3;
            copy16_g2l(A + (long long)(m0 + row) * lda + (k0 + col),
                       &As[buf][row * LDSS + col]);
        }
        if (B_NT) {
            // B tile as Bs[n][k], rows of B are K-contiguous: 1 chunk/thread
            const int row = tid >> 2;
            const int col = (tid & 3) << 3;
            copy16_g2l(Bm + (long long)(n0 + row) * ldb + (k0 + col),
                       &Bs[buf][row * LDSS + col]);
        } else {
            // B is [K,N]: coalesced read along N, transpose-scatter into LDS
            const int krow = tid >> 3;
            const int ncol = (tid & 7) << 3;
            v8bf v = *(const v8bf*)(Bm + (long long)(k0 + krow) * ldb + (n0 + ncol));
            #pragma unroll
            for (int e = 0; e < 8; ++e) Bs[buf][(ncol + e) * LDSS + krow] = v[e];
            if (k0 + 2 * TLK < K)   // gfx1250 global_prefetch for tile after next
                __builtin_prefetch(Bm + (long long)(k0 + 2 * TLK + krow) * ldb + (n0 + ncol), 0, 0);
        }
    };

    stage(0, 0);
    int buf = 0;
    for (int k0 = 0; k0 < K; k0 += TLK) {
        async_join();          // this wave's async copies into `buf` complete
        __syncthreads();       // everyone's staging done; prior reads retired
        if (k0 + TLK < K) stage(buf ^ 1, k0 + TLK);   // overlaps with WMMAs

        v16bf a0 = frag_a(As[buf], wm,      lane);
        v16bf a1 = frag_a(As[buf], wm + 16, lane);
        v16bf b0 = frag_b(Bs[buf], wn,      lane);
        v16bf b1 = frag_b(Bs[buf], wn + 16, lane);
        acc[0][0] = __builtin_amdgcn_wmma_f32_16x16x32_bf16(false, a0, false, b0, (short)0, acc[0][0], false, false);
        acc[0][1] = __builtin_amdgcn_wmma_f32_16x16x32_bf16(false, a0, false, b1, (short)0, acc[0][1], false, false);
        acc[1][0] = __builtin_amdgcn_wmma_f32_16x16x32_bf16(false, a1, false, b0, (short)0, acc[1][0], false, false);
        acc[1][1] = __builtin_amdgcn_wmma_f32_16x16x32_bf16(false, a1, false, b1, (short)0, acc[1][1], false, false);
        buf ^= 1;
    }

    // ---- epilogue: C/D layout — VGPR i, lanes<16: M=i,N=lane; lanes>=16: M=i+8
    const long long coff = (long long)zo * sCo + (long long)zi * sCi;
    const int hi2 = lane >> 4;
    const int lc  = lane & 15;
    #pragma unroll
    for (int i = 0; i < 2; ++i) {
        #pragma unroll
        for (int j = 0; j < 2; ++j) {
            #pragma unroll
            for (int e = 0; e < 8; ++e) {
                const int r = m0 + wm + i * 16 + e + 8 * hi2;
                const int c = n0 + wn + j * 16 + lc;
                const float v = acc[i][j][e] * scale;
                if (OUT_BF16)
                    ((bf16_t*)Cout)[coff + (long long)r * ldc + c] = (bf16_t)v;
                else
                    ((float*)Cout)[coff + (long long)r * ldc + c] = v;
            }
        }
    }
}

// ---------------------------------------------------------------------------
// Column softmax over L (axis=-2) -> pAlpha [B,R,L] bf16.  grid (R, B), 256 thr
// ---------------------------------------------------------------------------
__global__ __launch_bounds__(256) void softmax_col_kernel(
    const float* __restrict__ pScore, const unsigned char* __restrict__ mask,
    bf16_t* __restrict__ pAlpha)
{
    __shared__ float red[256];
    const int r = blockIdx.x, b = blockIdx.y, t = threadIdx.x;
    const float* col = pScore + (long long)b * cL * cR + r;
    const unsigned char* mrow = mask + (long long)b * cL * cL; // maskPAD[b,0,:]
    float v[cL / 256];
    float mx = -3.0e38f;
    #pragma unroll
    for (int i = 0; i < cL / 256; ++i) {
        const int l = t + i * 256;
        const float x = mrow[l] ? col[(long long)l * cR] : MASK_NEG_F;
        v[i] = x; mx = fmaxf(mx, x);
    }
    mx = block_max256(mx, red);
    float s = 0.0f;
    #pragma unroll
    for (int i = 0; i < cL / 256; ++i) { v[i] = __expf(v[i] - mx); s += v[i]; }
    s = block_sum256(s, red);
    const float inv = 1.0f / s;
    bf16_t* out = pAlpha + (long long)b * cR * cL + (long long)r * cL;
    #pragma unroll
    for (int i = 0; i < cL / 256; ++i) out[t + i * 256] = (bf16_t)(v[i] * inv);
}

// ---------------------------------------------------------------------------
// Row softmax over R (axis=-1) -> pAlpha_ [B,L,R] and pAlphaT [B,R,L] (bf16).
// grid (L, B), 256 threads (R == 256).
// ---------------------------------------------------------------------------
__global__ __launch_bounds__(256) void softmax_row_kernel(
    const float* __restrict__ pScore, const unsigned char* __restrict__ mask,
    bf16_t* __restrict__ pAlpha_, bf16_t* __restrict__ pAlphaT)
{
    __shared__ float red[256];
    const int l = blockIdx.x, b = blockIdx.y, t = threadIdx.x;
    const unsigned char m = mask[(long long)b * cL * cL + l];
    const float x = m ? pScore[(long long)b * cL * cR + (long long)l * cR + t]
                      : MASK_NEG_F;
    const float mx = block_max256(x, red);
    const float e  = __expf(x - mx);
    const float s  = block_sum256(e, red);
    const float a  = e / s;
    pAlpha_[(long long)b * cL * cR + (long long)l * cR + t] = (bf16_t)a;
    pAlphaT[(long long)b * cR * cL + (long long)t * cL + l] = (bf16_t)a;
}

// ---------------------------------------------------------------------------
// Softmax over attention scores rows -> alpha [B*H,R,R] bf16.
// Also emits per-row loss contribution (1 - alpha[r,r]) * inv.  grid (R, B*H).
// ---------------------------------------------------------------------------
__global__ __launch_bounds__(256) void softmax_alpha_kernel(
    const float* __restrict__ scores, bf16_t* __restrict__ alpha,
    float* __restrict__ rowpart, float inv)
{
    __shared__ float red[256];
    const int r = blockIdx.x, bh = blockIdx.y, t = threadIdx.x;
    const long long base = ((long long)bh * cR + r) * cR;
    const float x = scores[base + t];
    const float mx = block_max256(x, red);
    const float e  = __expf(x - mx);
    const float s  = block_sum256(e, red);
    const float a  = e / s;
    alpha[base + t] = (bf16_t)a;
    if (t == r) rowpart[(long long)bh * cR + r] = (1.0f - a) * inv;
}

// ---------------------------------------------------------------------------
// Off-diagonal squared sum partials over G [B,R,R].  grid = B*R*R/256 blocks.
// ---------------------------------------------------------------------------
__global__ __launch_bounds__(256) void gram_loss_partial_kernel(
    const float* __restrict__ G, float* __restrict__ part)
{
    __shared__ float red[256];
    const long long idx = (long long)blockIdx.x * 256 + threadIdx.x;
    const int rs = (int)(idx & (long long)(cR * cR - 1));
    const int r = rs >> 8, s = rs & 255;
    const float g = (r == s) ? 0.0f : G[idx];
    const float sum = block_sum256(g * g, red);
    if (threadIdx.x == 0) part[blockIdx.x] = sum;
}

// Single-block deterministic reduction: *dst += scale * sum(part[0..n))
__global__ __launch_bounds__(256) void reduce_add_kernel(
    const float* __restrict__ part, int n, float scale, float* __restrict__ dst)
{
    __shared__ float red[256];
    float s = 0.0f;
    for (int i = threadIdx.x; i < n; i += 256) s += part[i];
    s = block_sum256(s, red);
    if (threadIdx.x == 0) *dst += s * scale;
}

__global__ void zero_kernel(float* __restrict__ p, int n)
{
    const int i = blockIdx.x * blockDim.x + threadIdx.x;
    if (i < n) p[i] = 0.0f;
}

__global__ void finalize_loss_kernel(const float* __restrict__ addLoss,
                                     const float* __restrict__ acc,
                                     float* __restrict__ loss_out)
{
    if (threadIdx.x == 0) loss_out[0] = addLoss[0] + acc[0] + acc[1];
}

// ---------------------------------------------------------------------------
// Host orchestration
// ---------------------------------------------------------------------------
extern "C" void kernel_launch(void* const* d_in, const int* in_sizes, int n_in,
                              void* d_out, int out_size, void* d_ws, size_t ws_size,
                              hipStream_t stream)
{
    (void)in_sizes; (void)n_in; (void)out_size; (void)ws_size;

    const float*         qx      = (const float*)d_in[0];
    const float*         kx      = (const float*)d_in[1];
    const float*         vx      = (const float*)d_in[2];
    const float*         addLoss = (const float*)d_in[3];
    const unsigned char* maskPAD = (const unsigned char*)d_in[4];
    const float*         Wp      = (const float*)d_in[5];
    const float*         WQ      = (const float*)d_in[6];
    const float*         WK      = (const float*)d_in[7];
    const float*         WV      = (const float*)d_in[8];
    const float*         WO      = (const float*)d_in[9];

    float* z_out      = (float*)d_out;                          // [B,L,C]
    float* scores_out = z_out + (size_t)cB * cL * cC;           // [B,H,R,R]
    float* loss_out   = scores_out + (size_t)cB * cH * cR * cR; // scalar

    // ---- workspace carve-up (256B aligned) ----
    char* wp = (char*)d_ws;
    auto alloc = [&](size_t bytes) {
        char* q = wp; wp += (bytes + 255) & ~(size_t)255; return q;
    };
    const size_t nBLC = (size_t)cB * cL * cC;
    bf16_t* qxb = (bf16_t*)alloc(nBLC * 2);
    bf16_t* kxb = (bf16_t*)alloc(nBLC * 2);
    bf16_t* vxb = (bf16_t*)alloc(nBLC * 2);
    bf16_t* Wpb = (bf16_t*)alloc((size_t)cR * cC * 2);
    bf16_t* WQb = (bf16_t*)alloc((size_t)cHD * cC * 2);
    bf16_t* WKb = (bf16_t*)alloc((size_t)cHD * cC * 2);
    bf16_t* WVb = (bf16_t*)alloc((size_t)cHD * cC * 2);
    bf16_t* WOb = (bf16_t*)alloc((size_t)cC * cHD * 2);
    float*  pSc = (float*)alloc((size_t)cB * cL * cR * 4);      // pScore f32
    bf16_t* pA  = (bf16_t*)alloc((size_t)cB * cR * cL * 2);     // pAlpha  [B,R,L]
    bf16_t* pA_ = (bf16_t*)alloc((size_t)cB * cL * cR * 2);     // pAlpha_ [B,L,R]
    bf16_t* pAT = (bf16_t*)alloc((size_t)cB * cR * cL * 2);     // pAlpha_^T
    bf16_t* qr  = (bf16_t*)alloc((size_t)cB * cR * cC * 2);
    bf16_t* kr  = (bf16_t*)alloc((size_t)cB * cR * cC * 2);
    bf16_t* vr  = (bf16_t*)alloc((size_t)cB * cR * cC * 2);
    bf16_t* qh  = (bf16_t*)alloc((size_t)cB * cR * cHD * 2);
    bf16_t* kh  = (bf16_t*)alloc((size_t)cB * cR * cHD * 2);
    bf16_t* vh  = (bf16_t*)alloc((size_t)cB * cR * cHD * 2);
    float*  acc = (float*)alloc(256);
    float*  part = (float*)alloc((size_t)cB * cH * cR * 4);     // 16384 floats
    // alias the (consumed) pScore region for late-stage buffers:
    bf16_t* alpha = (bf16_t*)pSc;                               // 8.39 MB
    bf16_t* z1    = (bf16_t*)((char*)pSc + 8388608);            // [B,R,HD] 2.1MB
    bf16_t* z2    = (bf16_t*)((char*)pSc + 10485760);           // [B,R,C]  2.1MB
    float*  G     = (float*)((char*)pSc + 12582912);            // [B,R,R]  2.1MB

    const float invBRR  = 1.0f / (float)(cB * cR * cR);
    const float invBHRR = 1.0f / (float)(cB * cH * cR * cR);

    zero_kernel<<<1, 32, 0, stream>>>(acc, 4);

    // ---- f32 -> bf16 conversions ----
    auto cvt = [&](const float* in, bf16_t* out, long long n) {
        f32_to_bf16_kernel<<<(unsigned)((n / 4 + 255) / 256), 256, 0, stream>>>(in, out, n);
    };
    cvt(qx, qxb, (long long)nBLC);
    cvt(kx, kxb, (long long)nBLC);
    cvt(vx, vxb, (long long)nBLC);
    cvt(Wp, Wpb, (long long)cR * cC);
    cvt(WQ, WQb, (long long)cHD * cC);
    cvt(WK, WKb, (long long)cHD * cC);
    cvt(WV, WVb, (long long)cHD * cC);
    cvt(WO, WOb, (long long)cC * cHD);

    // ---- pScore[B,L,R] = vx @ Wp^T ----
    gemm_bf16_kernel<true, false><<<dim3(cL / TLM, cR / TLN, cB), 256, 0, stream>>>(
        vxb, Wpb, pSc, cC, cC, cC, cR,
        (long long)cL * cC, 0, 0, 0, (long long)cL * cR, 0, 1, 1.0f);

    // ---- softmaxes ----
    softmax_col_kernel<<<dim3(cR, cB), 256, 0, stream>>>(pSc, maskPAD, pA);
    softmax_row_kernel<<<dim3(cL, cB), 256, 0, stream>>>(pSc, maskPAD, pA_, pAT);

    // ---- gram1 = pAlpha @ pAlpha^T, loss term 1 ----
    gemm_bf16_kernel<true, false><<<dim3(cR / TLM, cR / TLN, cB), 256, 0, stream>>>(
        pA, pA, G, cL, cL, cL, cR,
        (long long)cR * cL, 0, (long long)cR * cL, 0, (long long)cR * cR, 0, 1, 1.0f);
    gram_loss_partial_kernel<<<(cB * cR * cR) / 256, 256, 0, stream>>>(G, part);
    reduce_add_kernel<<<1, 256, 0, stream>>>(part, (cB * cR * cR) / 256, invBRR, acc + 0);

    // ---- qr/kr/vr[B,R,C] = pAlpha @ {q,k,v}x ----
    const bf16_t* xs[3]  = {qxb, kxb, vxb};
    bf16_t*       rs_[3] = {qr, kr, vr};
    for (int i = 0; i < 3; ++i)
        gemm_bf16_kernel<false, true><<<dim3(cR / TLM, cC / TLN, cB), 256, 0, stream>>>(
            pA, xs[i], rs_[i], cL, cL, cC, cC,
            (long long)cR * cL, 0, (long long)cL * cC, 0, (long long)cR * cC, 0, 1, 1.0f);

    // ---- q/k/v head projections [B,R,HD] = {q,k,v}r @ W^T ----
    const bf16_t* Ws[3] = {WQb, WKb, WVb};
    bf16_t*       hs[3] = {qh, kh, vh};
    for (int i = 0; i < 3; ++i)
        gemm_bf16_kernel<true, true><<<dim3(cR / TLM, cHD / TLN, cB), 256, 0, stream>>>(
            rs_[i], Ws[i], hs[i], cC, cC, cC, cHD,
            (long long)cR * cC, 0, 0, 0, (long long)cR * cHD, 0, 1, 1.0f);

    // ---- scores[b,h,R,R] = Q K^T / sqrt(DK)  (per-head strided NT GEMM) ----
    gemm_bf16_kernel<true, false><<<dim3(cR / TLM, cR / TLN, cB * cH), 256, 0, stream>>>(
        qh, kh, scores_out, cDK, cHD, cHD, cR,
        (long long)cR * cHD, (long long)cDK,
        (long long)cR * cHD, (long long)cDK,
        (long long)cH * cR * cR, (long long)cR * cR, cH, 0.125f);

    // ---- alpha = softmax(scores), loss term 2 ----
    softmax_alpha_kernel<<<dim3(cR, cB * cH), 256, 0, stream>>>(
        scores_out, alpha, part, invBHRR);
    reduce_add_kernel<<<1, 256, 0, stream>>>(part, cB * cH * cR, 1.0f, acc + 1);

    // ---- z1[b, r, h*DK + d] = alpha @ V  (per-head strided NN GEMM) ----
    gemm_bf16_kernel<false, true><<<dim3(cR / TLM, cDK / TLN, cB * cH), 256, 0, stream>>>(
        alpha, vh, z1, cR, cR, cHD, cHD,
        (long long)cH * cR * cR, (long long)cR * cR,
        (long long)cR * cHD, (long long)cDK,
        (long long)cR * cHD, (long long)cDK, cH, 1.0f);

    // ---- z2[B,R,C] = z1 @ WO^T ----
    gemm_bf16_kernel<true, true><<<dim3(cR / TLM, cC / TLN, cB), 256, 0, stream>>>(
        z1, WOb, z2, cHD, cHD, cHD, cC,
        (long long)cR * cHD, 0, 0, 0, (long long)cR * cC, 0, 1, 1.0f);

    // ---- gram3 = pAlpha_^T @ pAlpha_, loss term 3 ----
    gemm_bf16_kernel<true, false><<<dim3(cR / TLM, cR / TLN, cB), 256, 0, stream>>>(
        pAT, pAT, G, cL, cL, cL, cR,
        (long long)cR * cL, 0, (long long)cR * cL, 0, (long long)cR * cR, 0, 1, 1.0f);
    gram_loss_partial_kernel<<<(cB * cR * cR) / 256, 256, 0, stream>>>(G, part);
    reduce_add_kernel<<<1, 256, 0, stream>>>(part, (cB * cR * cR) / 256, invBRR, acc + 0);

    // ---- z[B,L,C] = pAlpha_ @ z2 ----
    gemm_bf16_kernel<false, false><<<dim3(cL / TLM, cC / TLN, cB), 256, 0, stream>>>(
        pA_, z2, z_out, cR, cR, cC, cC,
        (long long)cL * cR, 0, (long long)cR * cC, 0, (long long)cL * cC, 0, 1, 1.0f);

    // ---- addLoss ----
    finalize_loss_kernel<<<1, 32, 0, stream>>>(addLoss, acc, loss_out);
}
